// CoeusBlock_13297218748577
// MI455X (gfx1250) — matrix operand order repaired and checked by reading, not compiled, not run on hardware
//
#include <hip/hip_runtime.h>
#include <hip/hip_bf16.h>
#include <stdint.h>

// ---------------------------------------------------------------------------
// Problem constants (from reference)
// ---------------------------------------------------------------------------
constexpr int Bc  = 2;
constexpr int Tc  = 2048;
constexpr int Dc  = 1024;
constexpr int Hc  = 16;
constexpr int KVc = 4;
constexpr int HDc = 64;
constexpr int HIDc = 3072;
constexpr int Mrows = Bc * Tc;           // 4096

typedef __attribute__((ext_vector_type(16))) __bf16 v16bf;
typedef __attribute__((ext_vector_type(8)))  float  v8f;
typedef __attribute__((ext_vector_type(4)))  unsigned int u32x4;
typedef __attribute__((ext_vector_type(8)))  int          i32x8;
typedef __attribute__((ext_vector_type(4)))  int          i32x4;

#ifndef __has_builtin
#define __has_builtin(x) 0
#endif
#if defined(__gfx1250__) && __has_builtin(__builtin_amdgcn_tensor_load_to_lds) \
                         && __has_builtin(__builtin_amdgcn_s_wait_tensorcnt)
#define HAS_TDM 1
#else
#define HAS_TDM 0
#endif

// fp32 -> bf16 round-to-nearest-even (bit twiddle; memory type is ushort)
__device__ __forceinline__ unsigned short f2bf(float x) {
  unsigned int u = __float_as_uint(x);
  unsigned int r = u + 0x7FFFu + ((u >> 16) & 1u);
  return (unsigned short)(r >> 16);
}

// two 16-byte loads -> one 32-byte bf16 fragment
__device__ __forceinline__ v16bf ldfrag(const unsigned short* p0,
                                        const unsigned short* p1) {
  union { v16bf v; uint4 q[2]; } u;
  u.q[0] = *(const uint4*)p0;
  u.q[1] = *(const uint4*)p1;
  return u.v;
}

__device__ __forceinline__ v8f wmma_bf16(v16bf a, v16bf b, v8f c) {
  return __builtin_amdgcn_wmma_f32_16x16x32_bf16(false, a, false, b,
                                                 (short)0, c, false, false);
}

__device__ __forceinline__ float siluf(float x) {
  return x / (1.f + __expf(-x));
}

__device__ __forceinline__ unsigned ldsoff(const void* p) {
  // flat LDS aperture: addr[31:0] is the workgroup-relative LDS byte offset
  return (unsigned)(unsigned long long)(uintptr_t)p;
}

#if HAS_TDM
// Issue one TDM 2-D tile load: rows x 32 bf16 elements, row stride lda
// elements, into LDS at lds_off with 16B padding after each 64B row
// (pad_interval = 16 DWORDs -> enc 3, pad_amount = 4 DWORDs -> enc 3),
// matching the GS=40-half padded LDS layout used by the WMMA fragments.
__device__ __forceinline__ void tdm_load_tile(unsigned lds_off, const void* g,
                                              int lda_elems, int rows) {
  unsigned long long ga = (unsigned long long)(uintptr_t)g;
  u32x4 g0;
  g0[0] = 1u;                                      // count=1, user desc
  g0[1] = lds_off;                                 // lds_addr (bytes)
  g0[2] = (unsigned)(ga & 0xFFFFFFFFu);            // global_addr[31:0]
  g0[3] = (unsigned)((ga >> 32) & 0x01FFFFFFu)     // global_addr[56:32]
        | (2u << 30);                              // type=2 ("image")
  i32x8 g1;
  g1[0] = (1 << 16)        // data_size = 1 -> 2 bytes
        | (1 << 20)        // pad_enable
        | (3 << 22)        // pad_interval: 16 DWORDs
        | (3 << 25);       // pad_amount:   4 DWORDs
  g1[1] = (32 << 16);      // tensor_dim0[15:0] = 32 (bits 63:48)
  g1[2] = (rows << 16);    // tensor_dim1[15:0] = rows (bits 111:96? -> 95:80 lo)
  g1[3] = (32 << 16);      // tile_dim0 = 32 (bits 127:112)
  g1[4] = rows;            // tile_dim1 (bits 143:128)
  g1[5] = lda_elems;       // tensor_dim0_stride[31:0]
  g1[6] = 0;
  g1[7] = 0;
  i32x4 z4 = {0, 0, 0, 0};
#if __clang_major__ >= 23
  i32x8 z8 = {0, 0, 0, 0, 0, 0, 0, 0};
  __builtin_amdgcn_tensor_load_to_lds(g0, g1, z4, z4, z8, 0);
#else
  __builtin_amdgcn_tensor_load_to_lds(g0, g1, z4, z4, 0);
#endif
}
#endif

// ---------------------------------------------------------------------------
// Generic bf16 WMMA GEMM:  C[M,N] = A[M,K] (bf16, row major, lda)
//                                  x Bt[N,K]^T (bf16, "B transposed", ldb)
// 128x64x32 block tile, 256 threads = 8 waves (4M x 2N), wave tile 32x32
// (4 WMMAs / 4 fragment loads per K-step).  A/B tiles staged to LDS by the
// Tensor Data Mover (double buffered, TENSORcnt-pipelined); vector-load
// fallback if the TDM builtin is unavailable.
// Epilogue modes:
//   0: C = acc (f32)           1: C = acc f32 AND Cb = bf16(acc)
//   2: Cb = bf16(acc) only     3: C += acc (f32 accumulate)
//   4: Cb = bf16( silu(other[idx]) * acc )      (fused SwiGLU)
// Requires M%128==0, N%64==0, K%32==0 (true for every call here).
// ---------------------------------------------------------------------------
#define GS 40   // LDS row stride (halves): 80B rows, 16B aligned chunks

__global__ __launch_bounds__(256)
void gemm_bf16(const unsigned short* __restrict__ A, int lda,
               const unsigned short* __restrict__ Bt, int ldb,
               float* __restrict__ C, unsigned short* __restrict__ Cb, int ldc,
               const float* __restrict__ other, int K, int mode) {
  __shared__ unsigned short As[2][128 * GS];
  __shared__ unsigned short Bs[2][64 * GS];

  const int tid  = threadIdx.x;
  const int lane = tid & 31;
  const int wid  = tid >> 5;
  const int wm   = wid >> 1;     // 0..3  (M strip of 32)
  const int wn   = wid & 1;      // 0..1  (N strip of 32)
  const int bm   = blockIdx.y * 128;
  const int bn   = blockIdx.x * 64;

  const int ln15    = lane & 15;
  const int kof     = (lane >> 4) * 8;    // A-fragment K sub-offset
  const int khalf16 = (lane >> 4) * 16;   // unused (B fragments also use kof)

  v8f acc[2][2];
  const v8f vz = {0.f,0.f,0.f,0.f,0.f,0.f,0.f,0.f};
#pragma unroll
  for (int mi = 0; mi < 2; mi++)
#pragma unroll
    for (int ni = 0; ni < 2; ni++) acc[mi][ni] = vz;

  const int nt = K / 32;

#if HAS_TDM
  if (wid == 0) {   // prologue: stage tile 0 into buffer 0
    tdm_load_tile(ldsoff(&As[0][0]), &A[(size_t)bm * lda], lda, 128);
    tdm_load_tile(ldsoff(&Bs[0][0]), &Bt[(size_t)bn * ldb], ldb, 64);
  }
#endif

  for (int i = 0; i < nt; i++) {
    const int cur = i & 1;
#if HAS_TDM
    if (wid == 0) {
      // stage next tile into the other buffer (re-issue last tile as a
      // harmless dummy so the TENSORcnt wait value stays a constant)
      const int nx = (i + 1 < nt) ? (i + 1) : i;
      const int nb = (i + 1) & 1;
      tdm_load_tile(ldsoff(&As[nb][0]), &A[(size_t)bm * lda + nx * 32], lda, 128);
      tdm_load_tile(ldsoff(&Bs[nb][0]), &Bt[(size_t)bn * ldb + nx * 32], ldb, 64);
      __builtin_amdgcn_s_wait_tensorcnt(2);  // oldest stage (cur) complete
    }
    __syncthreads();
#else
    {
      const int ar = tid >> 1;          // 0..127
      const int ac = (tid & 1) * 16;
      *(uint4*)&As[cur][ar * GS + ac] =
          *(const uint4*)&A[(size_t)(bm + ar) * lda + i * 32 + ac];
      *(uint4*)&As[cur][ar * GS + ac + 8] =
          *(const uint4*)&A[(size_t)(bm + ar) * lda + i * 32 + ac + 8];
      const int br = tid >> 2;          // 0..63
      const int bc = (tid & 3) * 8;
      *(uint4*)&Bs[cur][br * GS + bc] =
          *(const uint4*)&Bt[(size_t)(bn + br) * ldb + i * 32 + bc];
      if (i + 1 < nt) {
        __builtin_prefetch(&A[(size_t)(bm + ar) * lda + i * 32 + 32 + ac], 0, 1);
        __builtin_prefetch(&Bt[(size_t)(bn + br) * ldb + i * 32 + 32 + bc], 0, 1);
      }
    }
    __syncthreads();
#endif

    const int m0 = wm * 32 + ln15;
    const int m1 = m0 + 16;
    const int n0 = wn * 32 + ln15;
    const int n1 = n0 + 16;
    v16bf a0 = ldfrag(&As[cur][m0 * GS + kof], &As[cur][m0 * GS + 16 + kof]);
    v16bf a1 = ldfrag(&As[cur][m1 * GS + kof], &As[cur][m1 * GS + 16 + kof]);
    v16bf b0 = ldfrag(&Bs[cur][n0 * GS + kof], &Bs[cur][n0 * GS + 16 + kof]);
    v16bf b1 = ldfrag(&Bs[cur][n1 * GS + kof], &Bs[cur][n1 * GS + 16 + kof]);
    acc[0][0] = wmma_bf16(a0, b0, acc[0][0]);
    acc[0][1] = wmma_bf16(a0, b1, acc[0][1]);
    acc[1][0] = wmma_bf16(a1, b0, acc[1][0]);
    acc[1][1] = wmma_bf16(a1, b1, acc[1][1]);
    __syncthreads();   // all waves done with 'cur' before TDM may overwrite it
  }

  // epilogue: C/D layout -> row = r + 8*(lane>>4), col = lane&15
  const int half8 = (lane >> 4) * 8;
#pragma unroll
  for (int mi = 0; mi < 2; mi++) {
    const int rowbase = bm + wm * 32 + mi * 16 + half8;
#pragma unroll
    for (int ni = 0; ni < 2; ni++) {
      const int gn = bn + wn * 32 + ni * 16 + ln15;
#pragma unroll
      for (int r = 0; r < 8; r++) {
        size_t idx = (size_t)(rowbase + r) * ldc + gn;
        float v = acc[mi][ni][r];
        if (mode == 0)      { C[idx] = v; }
        else if (mode == 1) { C[idx] = v; Cb[idx] = f2bf(v); }
        else if (mode == 2) { Cb[idx] = f2bf(v); }
        else if (mode == 3) { C[idx] += v; }
        else                { Cb[idx] = f2bf(siluf(other[idx]) * v); }
      }
    }
  }
}

// ---------------------------------------------------------------------------
// Flash attention (causal GQA).  Q bf16 [B,H,T,64], K bf16 [B,KV,T,64],
// Vt bf16 [B,KV,64,T].  Out bf16 [B*T, H*64] (ready as GEMM A for @wo).
// Block = 128 threads (4 waves), q tile 64 rows, each wave a 16-row strip.
// ---------------------------------------------------------------------------
#define AP 72   // LDS row stride halves (144B rows)

__global__ __launch_bounds__(128)
void flash_attn(const unsigned short* __restrict__ Q,
                const unsigned short* __restrict__ Kr,
                const unsigned short* __restrict__ Vt,
                unsigned short* __restrict__ Out) {
  __shared__ unsigned short Ks[64 * AP];      // K tile  [t][d]
  __shared__ unsigned short Vs[64 * AP];      // Vt tile [d][t]
  __shared__ unsigned short Ps[4][16 * AP];   // per-wave P tile [16][64]

  const int bh = blockIdx.y;
  const int b  = bh / Hc;
  const int h  = bh % Hc;
  const int kvh = h / (Hc / KVc);
  const int qblk = blockIdx.x * 64;

  const int tid  = threadIdx.x;
  const int lane = tid & 31;
  const int wid  = tid >> 5;
  const int ln15    = lane & 15;
  const int kof     = (lane >> 4) * 8;
  const int khalf16 = (lane >> 4) * 16;
  const int half8   = (lane >> 4) * 8;
  const int qr0 = qblk + wid * 16;

  const unsigned short* qbase = Q  + ((size_t)(b * Hc  + h  ) * Tc) * 64;
  const unsigned short* kbase = Kr + ((size_t)(b * KVc + kvh) * Tc) * 64;
  const unsigned short* vbase = Vt + ((size_t)(b * KVc + kvh) * 64) * Tc;

  // Q fragments for this wave's 16 rows, two K-steps (d=0..31, 32..63)
  const unsigned short* qp = qbase + (size_t)(qr0 + ln15) * 64;
  v16bf qa0 = ldfrag(qp + kof,      qp + 16 + kof);
  v16bf qa1 = ldfrag(qp + 32 + kof, qp + 48 + kof);

  v8f o[4];
  const v8f vz = {0.f,0.f,0.f,0.f,0.f,0.f,0.f,0.f};
#pragma unroll
  for (int nt = 0; nt < 4; nt++) o[nt] = vz;
  float mr[8], lr_[8];
#pragma unroll
  for (int r = 0; r < 8; r++) { mr[r] = -1e30f; lr_[r] = 0.f; }

  const int nkv = blockIdx.x + 1;   // causal: tiles up to & incl. diagonal
  for (int kt = 0; kt < nkv; kt++) {
    const int kvb = kt * 64;
    // cooperative stage K and Vt tiles (each 64x64 bf16)
#pragma unroll
    for (int i = 0; i < 4; i++) {
      int e  = tid + i * 128;
      int rr = e >> 3;
      int cc = (e & 7) * 8;
      *(uint4*)&Ks[rr * AP + cc] =
          *(const uint4*)&kbase[(size_t)(kvb + rr) * 64 + cc];
      *(uint4*)&Vs[rr * AP + cc] =
          *(const uint4*)&vbase[(size_t)rr * Tc + kvb + cc];
    }
    __syncthreads();

    // S = Q K^T (16x64), 8 WMMAs
    v8f s[4];
#pragma unroll
    for (int nt = 0; nt < 4; nt++) s[nt] = vz;
#pragma unroll
    for (int ks = 0; ks < 2; ks++) {
      v16bf qa = ks ? qa1 : qa0;
#pragma unroll
      for (int nt = 0; nt < 4; nt++) {
        const unsigned short* kb = &Ks[(nt * 16 + ln15) * AP + ks * 32 + kof];
        s[nt] = wmma_bf16(qa, ldfrag(kb, kb + 16), s[nt]);
      }
    }

    // scale + causal mask
#pragma unroll
    for (int nt = 0; nt < 4; nt++) {
      int gc = kvb + nt * 16 + ln15;
#pragma unroll
      for (int r = 0; r < 8; r++) {
        float v = s[nt][r] * 0.125f;       // 1/sqrt(64)
        int rowg = qr0 + r + half8;
        s[nt][r] = (gc > rowg) ? -1e30f : v;
      }
    }

    // online softmax; row lives across the 16 lanes of a half-wave
#pragma unroll
    for (int r = 0; r < 8; r++) {
      float mx = fmaxf(fmaxf(s[0][r], s[1][r]), fmaxf(s[2][r], s[3][r]));
#pragma unroll
      for (int off = 1; off < 16; off <<= 1)
        mx = fmaxf(mx, __shfl_xor(mx, off, 16));
      float mn = fmaxf(mr[r], mx);
      float sc = __expf(mr[r] - mn);
      mr[r] = mn;
      float rs = 0.f;
#pragma unroll
      for (int nt = 0; nt < 4; nt++) {
        float e = __expf(s[nt][r] - mn);
        s[nt][r] = e;
        rs += e;
      }
#pragma unroll
      for (int off = 1; off < 16; off <<= 1)
        rs += __shfl_xor(rs, off, 16);
      lr_[r] = lr_[r] * sc + rs;
#pragma unroll
      for (int nt = 0; nt < 4; nt++) o[nt][r] *= sc;
    }

    // stage P as bf16 into wave-private LDS, re-enter as A fragments
#pragma unroll
    for (int r = 0; r < 8; r++)
#pragma unroll
      for (int nt = 0; nt < 4; nt++)
        Ps[wid][(r + half8) * AP + nt * 16 + ln15] = f2bf(s[nt][r]);
    __syncthreads();

    // O += P V  (8 WMMAs)
#pragma unroll
    for (int ks = 0; ks < 2; ks++) {
      const unsigned short* pp = &Ps[wid][ln15 * AP + ks * 32];
      v16bf pa = ldfrag(pp + kof, pp + 16 + kof);
#pragma unroll
      for (int nt = 0; nt < 4; nt++) {
        const unsigned short* vb = &Vs[(nt * 16 + ln15) * AP + ks * 32 + kof];
        o[nt] = wmma_bf16(pa, ldfrag(vb, vb + 16), o[nt]);
      }
    }
    __syncthreads();
  }

  // write O / l  ->  [b*T + t][h*64 + d] bf16
#pragma unroll
  for (int r = 0; r < 8; r++) {
    float inv = 1.f / lr_[r];
    int tq = qr0 + r + half8;
    size_t ro = (size_t)(b * Tc + tq) * (Hc * 64) + h * 64;
#pragma unroll
    for (int nt = 0; nt < 4; nt++)
      Out[ro + nt * 16 + ln15] = f2bf(o[nt][r] * inv);
  }
}

// ---------------------------------------------------------------------------
// Elementwise / normalization / scan kernels
// ---------------------------------------------------------------------------
__global__ __launch_bounds__(256)
void wt_k(const float* __restrict__ w, unsigned short* __restrict__ out,
          int K, int Nsrc, size_t n) {  // out[n][k] = bf16(w[k][n])
  size_t i = (size_t)blockIdx.x * 256 + threadIdx.x;
  if (i >= n) return;
  size_t nIdx = i / K, k = i % K;
  out[i] = f2bf(w[k * (size_t)Nsrc + nIdx]);
}

__global__ __launch_bounds__(256)
void rmsnorm_k(const float* __restrict__ x, const float* __restrict__ w,
               float* __restrict__ outf, unsigned short* __restrict__ outb,
               float eps) {
  __shared__ float red[8];
  const int row = blockIdx.x;
  const float* xr = x + (size_t)row * Dc;
  float v[4]; float ss = 0.f;
#pragma unroll
  for (int i = 0; i < 4; i++) { v[i] = xr[i * 256 + threadIdx.x]; ss += v[i] * v[i]; }
  for (int off = 1; off < 32; off <<= 1) ss += __shfl_xor(ss, off, 32);
  int lane = threadIdx.x & 31, wid = threadIdx.x >> 5;
  if (lane == 0) red[wid] = ss;
  __syncthreads();
  if (threadIdx.x == 0) { float s = 0; for (int i = 0; i < 8; i++) s += red[i]; red[0] = s; }
  __syncthreads();
  float inv = rsqrtf(red[0] / (float)Dc + eps);
#pragma unroll
  for (int i = 0; i < 4; i++) {
    int c = i * 256 + threadIdx.x;
    float o = v[i] * inv * w[c];
    if (outf) outf[(size_t)row * Dc + c] = o;
    if (outb) outb[(size_t)row * Dc + c] = f2bf(o);
  }
}

__global__ __launch_bounds__(256)
void lnsilu_k(const float* __restrict__ g, const float* __restrict__ b1,
              const float* __restrict__ gam, const float* __restrict__ bet,
              unsigned short* __restrict__ outb) {
  __shared__ float red[16];
  const int row = blockIdx.x;
  float v[4]; float s = 0.f, s2 = 0.f;
#pragma unroll
  for (int i = 0; i < 4; i++) {
    int c = i * 256 + threadIdx.x;
    float t = g[(size_t)row * Dc + c] + b1[c];
    v[i] = t; s += t; s2 += t * t;
  }
  for (int off = 1; off < 32; off <<= 1) { s += __shfl_xor(s, off, 32); s2 += __shfl_xor(s2, off, 32); }
  int lane = threadIdx.x & 31, wid = threadIdx.x >> 5;
  if (lane == 0) { red[wid] = s; red[8 + wid] = s2; }
  __syncthreads();
  if (threadIdx.x == 0) {
    float a = 0, bq = 0;
    for (int i = 0; i < 8; i++) { a += red[i]; bq += red[8 + i]; }
    red[0] = a; red[1] = bq;
  }
  __syncthreads();
  float mu = red[0] / (float)Dc;
  float var = red[1] / (float)Dc - mu * mu;
  float inv = rsqrtf(var + 1e-5f);
#pragma unroll
  for (int i = 0; i < 4; i++) {
    int c = i * 256 + threadIdx.x;
    float gn = (v[i] - mu) * inv * gam[c] + bet[c];
    outb[(size_t)row * Dc + c] = f2bf(siluf(gn));
  }
}

__global__ __launch_bounds__(256)
void logsig_k(const float* __restrict__ glin, const float* __restrict__ gb,
              float* __restrict__ lf, size_t n) {
  size_t i = (size_t)blockIdx.x * 256 + threadIdx.x;
  if (i >= n) return;
  int c = (int)(i % Dc);
  float x = glin[i] + gb[c];
  lf[i] = fminf(x, 0.f) - log1pf(__expf(-fabsf(x)));
}

// sequential log-space scan: a=cumsum(log_f); L=cumlogsumexp(u-a); h=exp(a+L)
__global__ __launch_bounds__(256)
void scan_k(const float* __restrict__ lf, const float* __restrict__ u,
            float* __restrict__ hs) {
  int g = blockIdx.x * 256 + threadIdx.x;   // < B*D
  int b = g / Dc, d = g % Dc;
  float a = 0.f, L = -1e30f;
  for (int t = 0; t < Tc; t++) {
    size_t idx = ((size_t)(b * Tc + t)) * Dc + d;
    a += lf[idx];
    float xv = u[idx] - a;
    float mx = fmaxf(L, xv), mn = fminf(L, xv);
    L = mx + log1pf(__expf(mn - mx));
    hs[idx] = __expf(a + L);
  }
}

__global__ __launch_bounds__(256)
void rope_k(const float* __restrict__ lin, const float* __restrict__ cs,
            const float* __restrict__ sn, unsigned short* __restrict__ out,
            int NH, size_t n) {
  size_t i = (size_t)blockIdx.x * 256 + threadIdx.x;
  if (i >= n) return;
  int j = (int)(i & 31);
  size_t u2 = i >> 5;
  int h = (int)(u2 % NH); size_t u3 = u2 / NH;
  int t = (int)(u3 % Tc); int b = (int)(u3 / Tc);
  const float* p = lin + ((size_t)(b * Tc + t)) * (NH * 64) + h * 64;
  float qr = p[2 * j], qi = p[2 * j + 1];
  float c = cs[t * 32 + j], s = sn[t * 32 + j];
  size_t ob = (((size_t)(b * NH + h)) * Tc + t) * 64;
  out[ob + 2 * j]     = f2bf(qr * c - qi * s);
  out[ob + 2 * j + 1] = f2bf(qr * s + qi * c);
}

__global__ __launch_bounds__(256)
void vtr_k(const float* __restrict__ vlin, unsigned short* __restrict__ out,
           size_t n) {  // out [B,KV,64,T] = v_lin [B*T, KV*64]
  size_t i = (size_t)blockIdx.x * 256 + threadIdx.x;
  if (i >= n) return;
  int t = (int)(i % Tc); size_t u = i / Tc;
  int d = (int)(u % 64); size_t u2 = u / 64;
  int kv = (int)(u2 % KVc); int b = (int)(u2 / KVc);
  out[i] = f2bf(vlin[((size_t)(b * Tc + t)) * (KVc * 64) + kv * 64 + d]);
}

__global__ __launch_bounds__(256)
void mix_k(const float* __restrict__ glin2, const float* __restrict__ b2,
           const float* __restrict__ outg, const float* __restrict__ outl,
           const float* __restrict__ x, float* __restrict__ xnew, size_t n) {
  size_t i = (size_t)blockIdx.x * 256 + threadIdx.x;
  if (i >= n) return;
  int c = (int)(i % Dc);
  float gl = glin2[i] + b2[c];
  float gt = 1.f / (1.f + __expf(-gl));
  xnew[i] = x[i] + (1.f - gt) * outg[i] + gt * outl[i];
}

__global__ __launch_bounds__(256)
void add_k(const float* __restrict__ a, const float* __restrict__ b,
           float* __restrict__ o, size_t n) {
  size_t i = (size_t)blockIdx.x * 256 + threadIdx.x;
  if (i >= n) return;
  o[i] = a[i] + b[i];
}

// ---------------------------------------------------------------------------
// Host orchestration
// ---------------------------------------------------------------------------
static inline int NG(size_t n) { return (int)((n + 255) / 256); }

extern "C" void kernel_launch(void* const* d_in, const int* in_sizes, int n_in,
                              void* d_out, int out_size, void* d_ws, size_t ws_size,
                              hipStream_t stream) {
  (void)in_sizes; (void)n_in; (void)out_size; (void)ws_size;
  const float* xin   = (const float*)d_in[0];
  const float* fcos  = (const float*)d_in[1];
  const float* fsin  = (const float*)d_in[2];
  const float* anw   = (const float*)d_in[3];
  const float* wq    = (const float*)d_in[4];
  const float* wk    = (const float*)d_in[5];
  const float* wv    = (const float*)d_in[6];
  const float* wo    = (const float*)d_in[7];
  const float* gatew = (const float*)d_in[8];
  const float* gateb = (const float*)d_in[9];
  const float* pinw  = (const float*)d_in[10];
  const float* poutw = (const float*)d_in[11];
  const float* cnw   = (const float*)d_in[12];
  const float* lgrw  = (const float*)d_in[13];
  const float* lgaw  = (const float*)d_in[14];
  const float* lgw1  = (const float*)d_in[15];
  const float* lgb1  = (const float*)d_in[16];
  const float* lng   = (const float*)d_in[17];
  const float* lnb   = (const float*)d_in[18];
  const float* lgw2  = (const float*)d_in[19];
  const float* lgb2  = (const float*)d_in[20];
  const float* ffnw  = (const float*)d_in[21];
  const float* rsnw  = (const float*)d_in[22];
  const float* rsw1  = (const float*)d_in[23];
  const float* rsw2  = (const float*)d_in[24];
  const float* rsw3  = (const float*)d_in[25];

  // ---- workspace bump allocator ----
  size_t off = 0;
  char* base = (char*)d_ws;
  auto ALLOC = [&](size_t bytes) -> char* {
    char* p = base + off;
    off += (bytes + 255) & ~(size_t)255;
    return p;
  };
  const size_t MD  = (size_t)Mrows * Dc;         // 4M elements
  const size_t MH  = (size_t)Mrows * HIDc;       // 12M elements

  // bf16 transposed weights [N][K]
  unsigned short* wq_t  = (unsigned short*)ALLOC(1024ull * 1024 * 2);
  unsigned short* wk_t  = (unsigned short*)ALLOC(256ull * 1024 * 2);
  unsigned short* wv_t  = (unsigned short*)ALLOC(256ull * 1024 * 2);
  unsigned short* wo_t  = (unsigned short*)ALLOC(1024ull * 1024 * 2);
  unsigned short* gw_t  = (unsigned short*)ALLOC(1024ull * 1024 * 2);
  unsigned short* pin_t = (unsigned short*)ALLOC(1024ull * 1024 * 2);   // u half only
  unsigned short* pou_t = (unsigned short*)ALLOC(1024ull * 1024 * 2);
  unsigned short* lgr_t = (unsigned short*)ALLOC(1024ull * 1024 * 2);
  unsigned short* lga_t = (unsigned short*)ALLOC(1024ull * 1024 * 2);
  unsigned short* lw1_t = (unsigned short*)ALLOC(1024ull * 2048 * 2);
  unsigned short* lw2_t = (unsigned short*)ALLOC(1024ull * 1024 * 2);
  unsigned short* r1_t  = (unsigned short*)ALLOC(3072ull * 1024 * 2);
  unsigned short* r2_t  = (unsigned short*)ALLOC(1024ull * 3072 * 2);
  unsigned short* r3_t  = (unsigned short*)ALLOC(3072ull * 1024 * 2);

  // activations
  unsigned short* h_bf     = (unsigned short*)ALLOC(MD * 2);
  float*          q_lin    = (float*)ALLOC(MD * 4);
  float*          k_lin    = (float*)ALLOC((size_t)Mrows * 256 * 4);
  float*          v_lin    = (float*)ALLOC((size_t)Mrows * 256 * 4);
  unsigned short* q_rope   = (unsigned short*)ALLOC(MD * 2);
  unsigned short* k_rope   = (unsigned short*)ALLOC((size_t)Bc * KVc * Tc * 64 * 2);
  unsigned short* vt_b     = (unsigned short*)ALLOC((size_t)Bc * KVc * 64 * Tc * 2);
  unsigned short* attn_bf  = (unsigned short*)ALLOC(MD * 2);
  float*          outl_f   = (float*)ALLOC(MD * 4);
  unsigned short* outl_bf  = (unsigned short*)ALLOC(MD * 2);
  float*          glin     = (float*)ALLOC(MD * 4);
  float*          logf_b   = (float*)ALLOC(MD * 4);
  float*          u_b      = (float*)ALLOC(MD * 4);
  float*          hstate   = (float*)ALLOC(MD * 4);
  unsigned short* hsn_bf   = (unsigned short*)ALLOC(MD * 2);
  float*          outg_f   = (float*)ALLOC(MD * 4);
  unsigned short* outg_bf  = (unsigned short*)ALLOC(MD * 2);
  unsigned short* hrnn_bf  = (unsigned short*)ALLOC(MD * 2);
  unsigned short* hattn_bf = (unsigned short*)ALLOC(MD * 2);
  float*          g_f      = (float*)ALLOC(MD * 4);
  unsigned short* gs_bf    = (unsigned short*)ALLOC(MD * 2);
  float*          glin2    = (float*)ALLOC(MD * 4);
  float*          xnew     = (float*)ALLOC(MD * 4);
  float*          state    = (float*)ALLOC(MD * 4);
  unsigned short* s_bf     = (unsigned short*)ALLOC(MD * 2);
  float*          t1       = (float*)ALLOC(MH * 4);
  unsigned short* ff_bf    = (unsigned short*)ALLOC(MH * 2);

  // ---- weight convert + transpose (fp32 -> bf16, [K,N] -> [N,K]) ----
  {
    size_t n;
    n = 1024ull * 1024; wt_k<<<NG(n), 256, 0, stream>>>(wq,   wq_t, 1024, 1024, n);
    n = 256ull  * 1024; wt_k<<<NG(n), 256, 0, stream>>>(wk,   wk_t, 1024, 256,  n);
    n = 256ull  * 1024; wt_k<<<NG(n), 256, 0, stream>>>(wv,   wv_t, 1024, 256,  n);
    n = 1024ull * 1024; wt_k<<<NG(n), 256, 0, stream>>>(wo,   wo_t, 1024, 1024, n);
    n = 1024ull * 1024; wt_k<<<NG(n), 256, 0, stream>>>(gatew,gw_t, 1024, 1024, n);
    n = 1024ull * 1024; wt_k<<<NG(n), 256, 0, stream>>>(pinw, pin_t,1024, 2048, n); // u cols only
    n = 1024ull * 1024; wt_k<<<NG(n), 256, 0, stream>>>(poutw,pou_t,1024, 1024, n);
    n = 1024ull * 1024; wt_k<<<NG(n), 256, 0, stream>>>(lgrw, lgr_t,1024, 1024, n);
    n = 1024ull * 1024; wt_k<<<NG(n), 256, 0, stream>>>(lgaw, lga_t,1024, 1024, n);
    n = 1024ull * 2048; wt_k<<<NG(n), 256, 0, stream>>>(lgw1, lw1_t,2048, 1024, n);
    n = 1024ull * 1024; wt_k<<<NG(n), 256, 0, stream>>>(lgw2, lw2_t,1024, 1024, n);
    n = 3072ull * 1024; wt_k<<<NG(n), 256, 0, stream>>>(rsw1, r1_t, 1024, 3072, n);
    n = 1024ull * 3072; wt_k<<<NG(n), 256, 0, stream>>>(rsw2, r2_t, 3072, 1024, n);
    n = 3072ull * 1024; wt_k<<<NG(n), 256, 0, stream>>>(rsw3, r3_t, 1024, 3072, n);
  }

  const dim3 blk(256);
  const dim3 gN1024(1024 / 64, Mrows / 128);
  const dim3 gN256 (256  / 64, Mrows / 128);
  const dim3 gN3072(3072 / 64, Mrows / 128);

  // 1) h = rms_norm(x, attn_norm_w)  -> bf16
  rmsnorm_k<<<Mrows, blk, 0, stream>>>(xin, anw, nullptr, h_bf, 1e-6f);

  // 2) QKV / gate / u projections
  gemm_bf16<<<gN1024, blk, 0, stream>>>(h_bf, 1024, wq_t, 1024, q_lin, nullptr, 1024, nullptr, 1024, 0);
  gemm_bf16<<<gN256,  blk, 0, stream>>>(h_bf, 1024, wk_t, 1024, k_lin, nullptr, 256,  nullptr, 1024, 0);
  gemm_bf16<<<gN256,  blk, 0, stream>>>(h_bf, 1024, wv_t, 1024, v_lin, nullptr, 256,  nullptr, 1024, 0);
  gemm_bf16<<<gN1024, blk, 0, stream>>>(h_bf, 1024, gw_t, 1024, glin,  nullptr, 1024, nullptr, 1024, 0);
  gemm_bf16<<<gN1024, blk, 0, stream>>>(h_bf, 1024, pin_t,1024, u_b,   nullptr, 1024, nullptr, 1024, 0);

  // 3) RoPE (q, k) + V transpose, bf16 attention layouts
  {
    size_t nq = (size_t)Mrows * Hc * 32;
    rope_k<<<NG(nq), blk, 0, stream>>>(q_lin, fcos, fsin, q_rope, Hc, nq);
    size_t nk = (size_t)Mrows * KVc * 32;
    rope_k<<<NG(nk), blk, 0, stream>>>(k_lin, fcos, fsin, k_rope, KVc, nk);
    size_t nv = (size_t)Bc * KVc * 64 * Tc;
    vtr_k<<<NG(nv), blk, 0, stream>>>(v_lin, vt_b, nv);
  }

  // 4) causal GQA flash attention
  flash_attn<<<dim3(Tc / 64, Bc * Hc), 128, 0, stream>>>(q_rope, k_rope, vt_b, attn_bf);

  // 5) out_local = attn @ wo   (f32 + bf16)
  gemm_bf16<<<gN1024, blk, 0, stream>>>(attn_bf, 1024, wo_t, 1024, outl_f, outl_bf, 1024, nullptr, 1024, 1);

  // 6) log_f = logsigmoid(h@gate_w + gate_b); scan; rms_norm(h_state)
  logsig_k<<<NG(MD), blk, 0, stream>>>(glin, gateb, logf_b, MD);
  scan_k<<<(Bc * Dc) / 256, blk, 0, stream>>>(logf_b, u_b, hstate);
  rmsnorm_k<<<Mrows, blk, 0, stream>>>(hstate, cnw, nullptr, hsn_bf, 1e-6f);

  // 7) out_global = rmsnorm(h_state) @ proj_out  (f32 + bf16)
  gemm_bf16<<<gN1024, blk, 0, stream>>>(hsn_bf, 1024, pou_t, 1024, outg_f, outg_bf, 1024, nullptr, 1024, 1);

  // 8) gate branch: h_rnn, h_attn, g = [h_rnn h_attn] @ lg_w1 (split-K accumulate)
  gemm_bf16<<<gN1024, blk, 0, stream>>>(outg_bf, 1024, lgr_t, 1024, nullptr, hrnn_bf,  1024, nullptr, 1024, 2);
  gemm_bf16<<<gN1024, blk, 0, stream>>>(outl_bf, 1024, lga_t, 1024, nullptr, hattn_bf, 1024, nullptr, 1024, 2);
  gemm_bf16<<<gN1024, blk, 0, stream>>>(hrnn_bf,  1024, lw1_t,        2048, g_f, nullptr, 1024, nullptr, 1024, 0);
  gemm_bf16<<<gN1024, blk, 0, stream>>>(hattn_bf, 1024, lw1_t + 1024, 2048, g_f, nullptr, 1024, nullptr, 1024, 3);

  // 9) layernorm(+b1) + silu -> bf16; gate = sigmoid(. @ lg_w2 + b2); mix + residual
  lnsilu_k<<<Mrows, blk, 0, stream>>>(g_f, lgb1, lng, lnb, gs_bf);
  gemm_bf16<<<gN1024, blk, 0, stream>>>(gs_bf, 1024, lw2_t, 1024, glin2, nullptr, 1024, nullptr, 1024, 0);
  mix_k<<<NG(MD), blk, 0, stream>>>(glin2, lgb2, outg_f, outl_f, xin, xnew, MD);

  // 10) residual FFN stack (DEPTH=2), SwiGLU fused in GEMM epilogue
  rmsnorm_k<<<Mrows, blk, 0, stream>>>(xnew, ffnw, state, nullptr, 1e-6f);
  for (int depth = 0; depth < 2; depth++) {
    rmsnorm_k<<<Mrows, blk, 0, stream>>>(state, rsnw, nullptr, s_bf, 1e-6f);
    gemm_bf16<<<gN3072, blk, 0, stream>>>(s_bf, 1024, r1_t, 1024, t1, nullptr, 3072, nullptr, 1024, 0);
    gemm_bf16<<<gN3072, blk, 0, stream>>>(s_bf, 1024, r3_t, 1024, nullptr, ff_bf, 3072, t1, 1024, 4);
    gemm_bf16<<<gN1024, blk, 0, stream>>>(ff_bf, 3072, r2_t, 3072, state, nullptr, 1024, nullptr, 3072, 3);
  }

  // 11) out = x_new + state
  add_k<<<NG(MD), blk, 0, stream>>>(xnew, state, (float*)d_out, MD);
}
